// SingleZPlanarNF2d_71141838291650
// MI455X (gfx1250) — matrix operand order repaired and checked by reading, not compiled, not run on hardware
//
#include <hip/hip_runtime.h>
#include <hip/hip_bf16.h>
#include <stdint.h>

typedef __attribute__((ext_vector_type(16))) _Float16     v16h;
typedef __attribute__((ext_vector_type(8)))  float        v8f;
typedef __attribute__((ext_vector_type(4)))  unsigned int u32x4;
typedef __attribute__((ext_vector_type(4)))  int          i32x4;
typedef __attribute__((ext_vector_type(8)))  int          i32x8;

// amdgpu-toolchain (clang-23 + therock-10.0 headers) ships this header and the
// 6-arg TDM builtin; ROCm 7.2 (clang-22) has the 5-arg form.
#if __has_include(<hip/amd_detail/amd_gfx1250_TDM.h>)
#define TDM_ARGS6 1
#else
#define TDM_ARGS6 0
#endif

namespace {
constexpr int Bn   = 32;
constexpr int Cn   = 256;
constexpr int Hn   = 64;
constexpr int Wn   = 64;
constexpr int HW   = Hn * Wn;        // 4096
constexpr int NPIX = Bn * HW;        // 131072
constexpr int TILE = 16;             // pixels per WMMA tile
constexpr int NTILES = NPIX / TILE;  // 8192
constexpr int ROWP = 17;             // padded LDS row stride in floats (TDM pad)
}

// Issue one TDM 2-D tile load: 256 channel-rows x 16 pixels (f32) from global
// into LDS, padding +1 DWORD after every 16 DWORDs (row stride -> 17 floats,
// bank-conflict-free half-wave reads). Tracked by TENSORcnt.
__device__ __forceinline__ void tdm_load_tile(const float* gsrc, unsigned int lds_byte_addr)
{
    const unsigned long long ga = (unsigned long long)(uintptr_t)gsrc;
    u32x4 g0;
    g0[0] = 1u;                                   // count=1 valid user descriptor
    g0[1] = lds_byte_addr;                        // lds_addr [63:32]
    g0[2] = (unsigned int)ga;                     // global_addr [95:64]
    g0[3] = (unsigned int)((ga >> 32) & 0x01FFFFFFu)  // global_addr [120:96]
          | (2u << 30);                           // type=2 ("image") [127:126]
    i32x8 g1;
    g1[0] = (int)((2u << 16)                      // data_size = 4 bytes
                | (1u << 20)                      // pad_enable
                | (3u << 22));                    // pad_interval: 8x8B = 16 DWORDs; pad_amount=0 -> 1 DWORD
    g1[1] = (int)(((unsigned)HW & 0xFFFFu) << 16);                  // tensor_dim0 lo16 (4096)
    g1[2] = (int)(((unsigned)HW >> 16) | (((unsigned)Cn) << 16));   // tensor_dim0 hi / tensor_dim1 lo (256)
    g1[3] = (int)(((unsigned)Cn >> 16) | (16u << 16));              // tensor_dim1 hi / tile_dim0 = 16 px
    g1[4] = Cn;                                   // tile_dim1 = 256 rows, tile_dim2 = 0
    g1[5] = HW;                                   // tensor_dim0_stride lo32 = 4096 elements
    g1[6] = 0;                                    // stride hi / tensor_dim1_stride lo
    g1[7] = 0;
    const i32x4 z4 = {0, 0, 0, 0};
#if TDM_ARGS6
    const i32x8 z8 = {0, 0, 0, 0, 0, 0, 0, 0};
    __builtin_amdgcn_tensor_load_to_lds(g0, g1, z4, z4, z8, 0);
#else
    __builtin_amdgcn_tensor_load_to_lds(g0, g1, z4, z4, 0);
#endif
}

__global__ __launch_bounds__(32, 1)
void planar_nf2d_tdm(const float* __restrict__ zin,
                     const float* __restrict__ h,
                     const float* __restrict__ Wu, const float* __restrict__ bu,
                     const float* __restrict__ Ww, const float* __restrict__ bw,
                     const float* __restrict__ Wb, const float* __restrict__ bb,
                     float* __restrict__ out)
{
    // One wave per workgroup: double-buffered h tile + u/w/b staging.
    __shared__ float hbuf[2][Cn * ROWP];   // 2 x 17408 B
    __shared__ float stg[3 * 16 * 16];     // 3 KB

    const int  lane = threadIdx.x & 31;
    const bool hi   = (lane >= 16);
    const int  lp   = lane & 15;
    const int  gwave = blockIdx.x;         // wave-uniform, scalar by construction
    const int  nwave = gridDim.x;

    // ---------------- Preload weight A-fragments (f32 -> f16) ----------------
    // A 16x32 f16 layout: lane row M = lp; halves 0..7 -> channels {0..7}+8*hi,
    // halves 8..15 -> channels {16..23}+8*hi within each 32-channel chunk.
    v16h aU[8], aW[8], aB[8];
#pragma unroll
    for (int cc = 0; cc < 8; ++cc) {
#pragma unroll
        for (int j = 0; j < 16; ++j) {
            const int c = cc * 32 + ((j < 8) ? j : j + 8) + (hi ? 8 : 0);
            aU[cc][j] = (_Float16)Wu[lp * Cn + c];
            aW[cc][j] = (_Float16)Ww[lp * Cn + c];
            aB[cc][j] = (_Float16)Wb[lp * Cn + c];
        }
    }
    // D-matrix element v at this lane is k = v + 8*hi.
    float bUv[8], bWv[8], bBv[8];
#pragma unroll
    for (int v = 0; v < 8; ++v) {
        const int k = v + (hi ? 8 : 0);
        bUv[v] = bu[k]; bWv[v] = bw[k]; bBv[v] = bb[k];
    }

    // LDS byte offsets of the two buffers (generic LDS addr: low 32 bits = offset).
    const unsigned int ldsb0 = (unsigned int)(uintptr_t)&hbuf[0][0];
    const unsigned int ldsb1 = (unsigned int)(uintptr_t)&hbuf[1][0];

    // Prologue: kick off the DMA for this wave's first tile.
    {
        const int n0 = gwave * TILE;
        tdm_load_tile(h + (size_t)(n0 / HW) * Cn * HW + (n0 % HW), ldsb0);
    }

    int cur = 0;
    for (int t = gwave; t < NTILES; t += nwave) {
        // Issue DMA for the next tile into the other buffer, then wait for the
        // current tile. DS reads of the other buffer from the previous
        // iteration must drain first (WAR vs the incoming DMA).
        const int tn = t + nwave;
        if (tn < NTILES) {
            asm volatile("s_wait_dscnt 0" ::: "memory");
            const int n1 = tn * TILE;
            tdm_load_tile(h + (size_t)(n1 / HW) * Cn * HW + (n1 % HW),
                          cur ? ldsb0 : ldsb1);
            __builtin_amdgcn_s_wait_tensorcnt(1);  // current tile complete
        } else {
            __builtin_amdgcn_s_wait_tensorcnt(0);
        }
        asm volatile("" ::: "memory");

        const float* hb = &hbuf[cur][0];

        v8f accU = {}, accW = {}, accB = {};
#pragma unroll
        for (int cc = 0; cc < 8; ++cc) {
            // B 32x16 f16: lane column N = lp; half j -> channel cb + (j<8 ? j : j+8).
            const int cb = cc * 32 + (hi ? 8 : 0);
            v16h bf;
#pragma unroll
            for (int j = 0; j < 16; ++j)
                bf[j] = (_Float16)hb[(cb + ((j < 8) ? j : j + 8)) * ROWP + lp];

            accU = __builtin_amdgcn_wmma_f32_16x16x32_f16(false, aU[cc], false, bf,
                                                          (short)0, accU, false, false);
            accW = __builtin_amdgcn_wmma_f32_16x16x32_f16(false, aW[cc], false, bf,
                                                          (short)0, accW, false, false);
            accB = __builtin_amdgcn_wmma_f32_16x16x32_f16(false, aB[cc], false, bf,
                                                          (short)0, accB, false, false);
        }

        // Bias + stage through LDS to get per-pixel (k-major) access.
#pragma unroll
        for (int v = 0; v < 8; ++v) {
            const int k = v + (hi ? 8 : 0);
            stg[0 * 256 + k * 16 + lp] = accU[v] + bUv[v];
            stg[1 * 256 + k * 16 + lp] = accW[v] + bWv[v];
            stg[2 * 256 + k * 16 + lp] = accB[v] + bBv[v];
        }
        asm volatile("s_wait_dscnt 0" ::: "memory");

        // ------------- Planar-flow scan: lanes 0..15, one pixel each -------------
        const int n0 = t * TILE;
        if (!hi) {
            float zv  = zin[n0 + lp];
            float ldj = 0.0f;
#pragma unroll
            for (int k = 0; k < 16; ++k) {
                const float u  = stg[0 * 256 + k * 16 + lp];
                const float w  = stg[1 * 256 + k * 16 + lp];
                const float bv = stg[2 * 256 + k * 16 + lp];
                const float uw = u * w;
                const float sp   = fmaxf(uw, 0.0f) + log1pf(__expf(-fabsf(uw)));
                const float uhat = u + (-1.0f + sp - uw) * (w / (w * w));
                const float tt   = tanhf(w * zv + bv);
                zv = zv + uhat * tt;
                const float psi = w * (1.0f - tt * tt);
                ldj += __logf(fabsf(1.0f + psi * uhat));
            }
            out[n0 + lp]        = zv;   // zk  (B,1,H,W)
            out[NPIX + n0 + lp] = ldj;  // ldj (B,1,H,W)
        }
        cur ^= 1;
    }
}

extern "C" void kernel_launch(void* const* d_in, const int* in_sizes, int n_in,
                              void* d_out, int out_size, void* d_ws, size_t ws_size,
                              hipStream_t stream) {
    (void)in_sizes; (void)n_in; (void)out_size; (void)d_ws; (void)ws_size;
    const float* z  = (const float*)d_in[0];
    const float* h  = (const float*)d_in[1];
    const float* Wu = (const float*)d_in[2];
    const float* bu = (const float*)d_in[3];
    const float* Ww = (const float*)d_in[4];
    const float* bw = (const float*)d_in[5];
    const float* Wb = (const float*)d_in[6];
    const float* bb = (const float*)d_in[7];
    float* out = (float*)d_out;

    // 2048 single-wave workgroups; 8192 tiles -> exactly 4 tiles per wave,
    // each tile's 16 KB h-slab DMA'd ahead by the TDM.
    planar_nf2d_tdm<<<2048, 32, 0, stream>>>(z, h, Wu, bu, Ww, bw, Wb, bb, out);
}
